// CharLSTM_35699768164953
// MI455X (gfx1250) — compile-verified, matrix operand order checked by reading, not run on hardware
//
#include <hip/hip_runtime.h>

// ---------------- problem constants ----------------
#define VSZ 128
#define H   1024
#define LNUM 2
#define BATCH 64
#define TSTEPS 512
#define G4  4096            // 4*H
#define NBLK 32             // persistent blocks (grid barrier participants)

typedef __attribute__((ext_vector_type(16))) __bf16 v16bf;
typedef __attribute__((ext_vector_type(8)))  __bf16 v8bf;
typedef __attribute__((ext_vector_type(8)))  float  v8f;

static __device__ __forceinline__ v16bf cat16(v8bf lo, v8bf hi) {
    return __builtin_shufflevector(lo, hi, 0,1,2,3,4,5,6,7,8,9,10,11,12,13,14,15);
}

static __device__ __forceinline__ v8f wmma_bf16(v16bf a, v16bf b, v8f c) {
    return __builtin_amdgcn_wmma_f32_16x16x32_bf16(
        /*neg_a=*/false, a, /*neg_b=*/false, b,
        /*c_mod=*/(short)0, c, /*reuse_a=*/false, /*reuse_b=*/false);
}

// ---- LDS A-tile helpers: 16 rows x 1024 cols bf16, XOR swizzle on 16B chunks ----
// chunk ci (8 bf16 = 16B) of row r lives at r*1024 + ((ci ^ (r&7)) << 3)
static __device__ __forceinline__ void stageRowF32(__bf16* lds, int r, int c0,
                                                   const float* __restrict__ src) {
#pragma unroll
    for (int ch = 0; ch < 8; ++ch) {
        int ci = (c0 >> 3) + ch;
        const float* s = src + ch * 8;
        v8bf v;
#pragma unroll
        for (int j = 0; j < 8; ++j) v[j] = (__bf16)s[j];
        *(v8bf*)(lds + r * 1024 + ((ci ^ (r & 7)) << 3)) = v;
    }
}
static __device__ __forceinline__ void stageRowBF(__bf16* lds, int r, int c0,
                                                  const __bf16* __restrict__ src) {
#pragma unroll
    for (int ch = 0; ch < 8; ++ch) {
        int ci = (c0 >> 3) + ch;
        *(v8bf*)(lds + r * 1024 + ((ci ^ (r & 7)) << 3)) = *(const v8bf*)(src + ch * 8);
    }
}
// A fragment (16x32 bf16): lane = kh*16 + row; elems e<8 -> K=kh*8+e, e>=8 -> K=16+kh*8+(e-8)
static __device__ __forceinline__ v16bf ldsAfrag(const __bf16* lds, int lane, int kt) {
    int r  = lane & 15;
    int kh = lane >> 4;
    int ci = kt * 4 + kh;
    v8bf lo = *(const v8bf*)(lds + r * 1024 + (((ci    ) ^ (r & 7)) << 3));
    v8bf hi = *(const v8bf*)(lds + r * 1024 + (((ci + 2) ^ (r & 7)) << 3));
    return cat16(lo, hi);
}

// ---- device-wide barrier: monotone arrive counter, zeroed by init kernel each launch ----
static __device__ __forceinline__ void gbar(unsigned* bar, unsigned& phase) {
    __syncthreads();
    ++phase;
    if (threadIdx.x == 0) {
        __hip_atomic_fetch_add(bar, 1u, __ATOMIC_RELEASE, __HIP_MEMORY_SCOPE_AGENT);
        const unsigned target = phase * (unsigned)NBLK;
        while (__hip_atomic_load(bar, __ATOMIC_ACQUIRE, __HIP_MEMORY_SCOPE_AGENT) < target)
            __builtin_amdgcn_s_sleep(1);
    }
    __syncthreads();
}

static __device__ __forceinline__ float sigmoidf_(float x) {
    return 1.0f / (1.0f + __expf(-x));
}

// ---------------- kernels ----------------
__global__ __launch_bounds__(256) void init_kernel(unsigned* bar, __bf16* hbuf, float* cbuf) {
    int i = blockIdx.x * 256 + threadIdx.x;
    if (i < 64) bar[i] = 0u;
    if (i < LNUM * BATCH * H) { hbuf[i] = (__bf16)0.0f; cbuf[i] = 0.0f; }
}

// repack fp32 row-major (K x Ncols) weights into bf16 WMMA-B fragment order:
// P[((ktg*NT + nt)*32 + lane)*16 + e] = W[k][n],
//   n = nt*16 + (lane&15),  k = ktg*32 + (e>>3)*16 + (lane>>4)*8 + (e&7)
__global__ __launch_bounds__(256) void pack_kernel(const float* __restrict__ W,
                                                   __bf16* __restrict__ P,
                                                   int Ncols, int NT, long total) {
    long i = (long)blockIdx.x * 256 + threadIdx.x;
    if (i >= total) return;
    int  e    = (int)(i & 15);
    int  lane = (int)((i >> 4) & 31);
    long fi   = i >> 9;
    int  nt   = (int)(fi % NT);
    int  ktg  = (int)(fi / NT);
    int  n    = nt * 16 + (lane & 15);
    int  k    = ktg * 32 + ((e >> 3) << 4) + ((lane >> 4) << 3) + (e & 7);
    P[i] = (__bf16)W[(long)k * Ncols + n];
}

// Persistent fused LSTM. 32 blocks = 4 M-tiles x 8 N-groups; each wave owns one
// 16-batch x 16-hcol tile for ALL FOUR gates (cols c, c+1024, c+2048, c+3072),
// so the cell update happens in registers: one device barrier per layer-step.
__global__ __launch_bounds__(256) void lstm_kernel(
    const int*   __restrict__ idx,    // (B,T)
    const float* __restrict__ embed,  // (V,H)
    const float* __restrict__ bias,   // (L,4H)
    const __bf16* __restrict__ WxP,   // packed (L,H,4H)
    const __bf16* __restrict__ WhP,   // packed (L,H,4H)
    __bf16* __restrict__ hbuf,        // (L,B,H) bf16
    float*  __restrict__ cbuf,        // (L,B,H) f32
    __bf16* __restrict__ hs,          // (T,B,H) bf16
    unsigned* __restrict__ bar)
{
    __shared__ __bf16 ldsX[16 * 1024];   // 32 KB
    __shared__ __bf16 ldsH[16 * 1024];   // 32 KB
    const int mt   = blockIdx.x >> 3;        // 0..3  batch 16-row tile
    const int ng   = blockIdx.x & 7;         // 0..7  n-group
    const int w    = threadIdx.x >> 5;       // wave 0..7
    const int lane = threadIdx.x & 31;
    const int cg   = ng * 8 + w;             // 0..63 h-column tile (16 cols)
    const int c0   = cg * 16;
    const int m0   = mt * 16;
    const int sr   = threadIdx.x >> 4;       // staging row 0..15
    const int sc   = (threadIdx.x & 15) * 64;
    unsigned phase = 0;

    for (int t = 0; t < TSTEPS; ++t) {
        for (int l = 0; l < LNUM; ++l) {
            // ---- stage A operands (16 rows x 1024 bf16) into LDS ----
            if (l == 0) {
                const int v = idx[(size_t)(m0 + sr) * TSTEPS + t];
                stageRowF32(ldsX, sr, sc, embed + (size_t)v * H + sc);
                // pull next timestep's embedding rows toward L2 (idx is known ahead)
                if (t + 1 < TSTEPS) {
                    const int v2 = idx[(size_t)(m0 + sr) * TSTEPS + t + 1];
                    __builtin_prefetch(embed + (size_t)v2 * H + sc, 0, 3);
                }
            } else {
                stageRowBF(ldsX, sr, sc, hbuf + (size_t)(m0 + sr) * H + sc); // layer0 h
            }
            stageRowBF(ldsH, sr, sc, hbuf + (size_t)l * (BATCH * H) + (size_t)(m0 + sr) * H + sc);
            __syncthreads();

            // ---- gate GEMM: four independent accumulator chains (i,f,g,o) ----
            v8f acc0 = {}, acc1 = {}, acc2 = {}, acc3 = {};
            // gate q occupies columns [q*1024, q*1024+1024) -> n-tile q*64 + cg
            const size_t fb = (size_t)(l * 32) * 256;   // fragment base for this layer
            const __bf16* wx0 = WxP + (fb + 0 * 64 + cg) * 512 + lane * 16;
            const __bf16* wx1 = WxP + (fb + 1 * 64 + cg) * 512 + lane * 16;
            const __bf16* wx2 = WxP + (fb + 2 * 64 + cg) * 512 + lane * 16;
            const __bf16* wx3 = WxP + (fb + 3 * 64 + cg) * 512 + lane * 16;
            const __bf16* wh0 = WhP + (fb + 0 * 64 + cg) * 512 + lane * 16;
            const __bf16* wh1 = WhP + (fb + 1 * 64 + cg) * 512 + lane * 16;
            const __bf16* wh2 = WhP + (fb + 2 * 64 + cg) * 512 + lane * 16;
            const __bf16* wh3 = WhP + (fb + 3 * 64 + cg) * 512 + lane * 16;
#pragma unroll 2
            for (int kt = 0; kt < 32; ++kt) {
                const size_t ko = (size_t)kt * (256 * 512);
                v16bf ax = ldsAfrag(ldsX, lane, kt);
                v16bf ah = ldsAfrag(ldsH, lane, kt);
                acc0 = wmma_bf16(ax, *(const v16bf*)(wx0 + ko), acc0);
                acc1 = wmma_bf16(ax, *(const v16bf*)(wx1 + ko), acc1);
                acc2 = wmma_bf16(ax, *(const v16bf*)(wx2 + ko), acc2);
                acc3 = wmma_bf16(ax, *(const v16bf*)(wx3 + ko), acc3);
                acc0 = wmma_bf16(ah, *(const v16bf*)(wh0 + ko), acc0);
                acc1 = wmma_bf16(ah, *(const v16bf*)(wh1 + ko), acc1);
                acc2 = wmma_bf16(ah, *(const v16bf*)(wh2 + ko), acc2);
                acc3 = wmma_bf16(ah, *(const v16bf*)(wh3 + ko), acc3);
            }

            // ---- in-register LSTM cell update (lane holds i,f,g,o of same element)
            {
                const int ncol = c0 + (lane & 15);
                const int mrow = m0 + ((lane >> 4) << 3);
                const float bi = bias[(size_t)l * G4 +        ncol];
                const float bf = bias[(size_t)l * G4 + 1024 + ncol];
                const float bg = bias[(size_t)l * G4 + 2048 + ncol];
                const float bo = bias[(size_t)l * G4 + 3072 + ncol];
#pragma unroll
                for (int r2 = 0; r2 < 8; ++r2) {
                    const int row = mrow + r2;
                    const size_t ei = (size_t)l * (BATCH * H) + (size_t)row * H + ncol;
                    const float ig = sigmoidf_(acc0[r2] + bi);
                    const float fg = sigmoidf_(acc1[r2] + bf);
                    const float gg = tanhf    (acc2[r2] + bg);
                    const float og = sigmoidf_(acc3[r2] + bo);
                    const float c  = fg * cbuf[ei] + ig * gg;
                    cbuf[ei] = c;
                    const float h = og * tanhf(c);
                    hbuf[ei] = (__bf16)h;
                    if (l == LNUM - 1)
                        hs[(size_t)t * (BATCH * H) + (size_t)row * H + ncol] = (__bf16)h;
                }
            }
            gbar(bar, phase);   // single device barrier per layer-step
        }
    }
}

// hs(32768 x 1024) @ W_out(1024 x 128) -> out (B,T,V) fp32
__global__ __launch_bounds__(256) void proj_kernel(const __bf16* __restrict__ hs,
                                                   const __bf16* __restrict__ WoP,
                                                   float* __restrict__ out) {
    const int w    = threadIdx.x >> 5;   // n-tile 0..7 (V=128)
    const int lane = threadIdx.x & 31;
    const int m0   = blockIdx.x * 16;    // rows r = t*B + b
    const int rowA = m0 + (lane & 15);
    const int kh   = lane >> 4;
    const __bf16* arow = hs + (size_t)rowA * H;
    v8f acc = {};
#pragma unroll 4
    for (int kt = 0; kt < 32; ++kt) {
        const int kb = kt * 32 + kh * 8;
        v8bf lo = *(const v8bf*)(arow + kb);
        v8bf hi = *(const v8bf*)(arow + kb + 16);
        v16bf a = cat16(lo, hi);
        v16bf b = *(const v16bf*)(WoP + ((size_t)(kt * 8 + w) * 32 + lane) * 16);
        acc = wmma_bf16(a, b, acc);
    }
    const int n  = w * 16 + (lane & 15);
    const int mB = m0 + ((lane >> 4) << 3);
#pragma unroll
    for (int r2 = 0; r2 < 8; ++r2) {
        const int row = mB + r2;
        const int tt = row >> 6;       // T index (B = 64)
        const int bb = row & 63;       // batch index
        out[((size_t)bb * TSTEPS + tt) * VSZ + n] = acc[r2];
    }
}

// ---------------- host launcher ----------------
extern "C" void kernel_launch(void* const* d_in, const int* in_sizes, int n_in,
                              void* d_out, int out_size, void* d_ws, size_t ws_size,
                              hipStream_t stream) {
    (void)in_sizes; (void)n_in; (void)out_size; (void)ws_size;
    const int*   idx   = (const int*)  d_in[0];
    const float* embed = (const float*)d_in[1];
    const float* Wx    = (const float*)d_in[2];
    const float* Wh    = (const float*)d_in[3];
    const float* bias  = (const float*)d_in[4];
    const float* Wo    = (const float*)d_in[5];
    float* out = (float*)d_out;
    char*  ws  = (char*)d_ws;

    // workspace layout (bytes): ~97 MB total
    const size_t SZ_WXP = (size_t)LNUM * H * G4 * 2;        // 16 MB
    const size_t SZ_WHP = SZ_WXP;                           // 16 MB
    const size_t SZ_WOP = (size_t)H * VSZ * 2;              // 256 KB
    const size_t SZ_HS  = (size_t)TSTEPS * BATCH * H * 2;   // 64 MB
    const size_t SZ_HB  = (size_t)LNUM * BATCH * H * 2;     // 256 KB
    const size_t SZ_CB  = (size_t)LNUM * BATCH * H * 4;     // 512 KB

    size_t off = 256;
    unsigned* bar  = (unsigned*)(ws + 0);
    __bf16*   WxP  = (__bf16*)(ws + off); off += SZ_WXP;
    __bf16*   WhP  = (__bf16*)(ws + off); off += SZ_WHP;
    __bf16*   WoP  = (__bf16*)(ws + off); off += SZ_WOP;
    __bf16*   hsB  = (__bf16*)(ws + off); off += SZ_HS;
    __bf16*   hbuf = (__bf16*)(ws + off); off += SZ_HB;
    float*    cbuf = (float*) (ws + off); off += SZ_CB;

    // 1) zero barrier + recurrent state (every launch -> graph-replay safe)
    init_kernel<<<512, 256, 0, stream>>>(bar, hbuf, cbuf);
    // 2) repack weights into bf16 WMMA fragment order (one 32B load per B-fragment per lane)
    pack_kernel<<<32768, 256, 0, stream>>>(Wx, WxP, G4, 256, (long)LNUM * H * G4);
    pack_kernel<<<32768, 256, 0, stream>>>(Wh, WhP, G4, 256, (long)LNUM * H * G4);
    pack_kernel<<<512,   256, 0, stream>>>(Wo, WoP, VSZ, 8,  (long)H * VSZ);
    // 3) persistent fused LSTM over T timesteps (one device barrier per layer-step)
    lstm_kernel<<<NBLK, 256, 0, stream>>>(idx, embed, bias, WxP, WhP,
                                          hbuf, cbuf, hsB, bar);
    // 4) output projection
    proj_kernel<<<(BATCH * TSTEPS) / 16, 256, 0, stream>>>(hsB, WoP, out);
}
